// MultiHeadAttention_43370579755636
// MI455X (gfx1250) — matrix operand order repaired and checked by reading, not compiled, not run on hardware
//
#include <hip/hip_runtime.h>

// ---------------------------------------------------------------------------
// Types for CDNA5 WMMA (wave32): D(16x16 f32) = A(16x32 bf16) x B(32x16 bf16) + C
// ---------------------------------------------------------------------------
typedef __bf16 bf16_t;
typedef __attribute__((ext_vector_type(16))) __bf16 v16bf;
typedef __attribute__((ext_vector_type(8)))  float  v8f;

#define B_  2
#define S_  2048
#define E_  1024
#define H_  16
#define D_  64
// q pre-scale: D^-0.5 * log2(e)  (softmax done in exp2 space -> bare v_exp_f32)
#define QSCALE_ 0.18033688011112042f
#define LOG2_10000 13.287712379549449f

union Frag16 { v16bf v; uint4 q[2]; uint u[8]; };
union Pack8  { uint4 q; uint u[4]; };

// A fragment (16x32, row-major src): lane holds row M = lane&15;
// 16 contiguous K values starting at k0 + 16*half
static __device__ inline v16bf load_a_frag(const bf16_t* __restrict__ row, int k0, int half) {
  Frag16 f;
  f.q[0] = *(const uint4*)(row + k0 + 16 * half);
  f.q[1] = *(const uint4*)(row + k0 + 16 * half + 8);
  return f.v;
}

// B fragment (32x16) from an (N,K)-row-major matrix: lane holds column N = lane&15,
// 16 contiguous K values starting at k0 + 16*half.
static __device__ inline v16bf load_b_frag(const bf16_t* __restrict__ nrow, int k0, int half) {
  Frag16 f;
  const uint4* p = (const uint4*)(nrow + k0 + 16 * half);
  f.q[0] = p[0];
  f.q[1] = p[1];
  return f.v;
}

static __device__ inline v8f wmma_bf16(v16bf a, v16bf b, v8f c) {
  return __builtin_amdgcn_wmma_f32_16x16x32_bf16(false, a, false, b, (short)0, c,
                                                 false, false);
}

static __device__ inline uint pack2_bf16(float a, float b) {
  union { bf16_t h[2]; uint u; } p;
  p.h[0] = (bf16_t)a;
  p.h[1] = (bf16_t)b;
  return p.u;
}

// Raw v_exp_f32: all softmax args are <= 0; underflow flushing is acceptable,
// so skip libm's range-fixup expansion (saves ~4 VALU per exponential).
static __device__ inline float fast_exp2(float x) {
  return __builtin_amdgcn_exp2f(x);
}

// ---------------------------------------------------------------------------
// Kernel 0: f32 -> bf16 conversion
// ---------------------------------------------------------------------------
__global__ void f32_to_bf16_k(const float* __restrict__ in, bf16_t* __restrict__ out, int n) {
  int i = blockIdx.x * blockDim.x + threadIdx.x;
  if (i < n) out[i] = (bf16_t)in[i];
}

// ---------------------------------------------------------------------------
// Kernels 1 & 4: C[M,N] = A[M,K] * B[N,K]^T.  One wave -> 16x64 output tile.
// ---------------------------------------------------------------------------
template <bool OUT_BF16>
__global__ __launch_bounds__(128) void gemm_bt_wmma(const bf16_t* __restrict__ A,
                                                    const bf16_t* __restrict__ Bm,
                                                    bf16_t* __restrict__ Cb,
                                                    float* __restrict__ Cf,
                                                    int M, int N, int K) {
  const int wid    = blockIdx.x * (blockDim.x >> 5) + (threadIdx.x >> 5);
  const int tilesN = N >> 6;
  const int tm     = wid / tilesN;
  const int tn     = wid - tm * tilesN;
  if ((tm << 4) >= M) return;                  // wave-uniform guard (EXEC stays full)

  const int lane = threadIdx.x & 31;
  const int half = lane >> 4;
  const int r    = lane & 15;
  const int m0   = tm << 4;
  const int n0   = tn << 6;

  v8f acc0 = {}, acc1 = {}, acc2 = {}, acc3 = {};

  const bf16_t* arow = A  + (size_t)(m0 + r) * K;
  const bf16_t* b0   = Bm + (size_t)(n0 + r) * K;
  const bf16_t* b1   = Bm + (size_t)(n0 + 16 + r) * K;
  const bf16_t* b2   = Bm + (size_t)(n0 + 32 + r) * K;
  const bf16_t* b3   = Bm + (size_t)(n0 + 48 + r) * K;

  for (int k0 = 0; k0 < K; k0 += 32) {
    v16bf a = load_a_frag(arow, k0, half);
    acc0 = wmma_bf16(a, load_b_frag(b0, k0, half), acc0);
    acc1 = wmma_bf16(a, load_b_frag(b1, k0, half), acc1);
    acc2 = wmma_bf16(a, load_b_frag(b2, k0, half), acc2);
    acc3 = wmma_bf16(a, load_b_frag(b3, k0, half), acc3);
  }

  // C/D layout: element e -> row m0 + e + 8*half, column n0 + (lane&15) (+16*t)
  const int col = n0 + r;
#pragma unroll
  for (int e = 0; e < 8; ++e) {
    const size_t row = (size_t)(m0 + e + 8 * half);
    if (OUT_BF16) {
      bf16_t* o = Cb + row * N + col;
      o[0]  = (bf16_t)acc0[e];
      o[16] = (bf16_t)acc1[e];
      o[32] = (bf16_t)acc2[e];
      o[48] = (bf16_t)acc3[e];
    } else {
      float* o = Cf + row * N + col;
      o[0]  = acc0[e];
      o[16] = acc1[e];
      o[32] = acc2[e];
      o[48] = acc3[e];
    }
  }
}

// ---------------------------------------------------------------------------
// Kernel 2: RoPE + head scatter.
//   qkv[b][s][f] bf16 -> q_rope (B,H,S,D) * QSCALE, k_rope (B,H,S,D), v_t (B,H,D,S)
// ---------------------------------------------------------------------------
__global__ void rope_scatter_k(const bf16_t* __restrict__ qkv,
                               bf16_t* __restrict__ qr,
                               bf16_t* __restrict__ kr,
                               bf16_t* __restrict__ vt) {
  const int i = blockIdx.x * blockDim.x + threadIdx.x;   // 2^21 threads exactly
  const int p = i & 31;                  // pair index (D/2 = 32)
  const int s = (i >> 5) & (S_ - 1);
  const int h = (i >> 16) & (H_ - 1);
  const int b = i >> 20;

  const size_t base = (size_t)(b * S_ + s) * (3 * E_) + h * D_ + 2 * p;
  const float inv = exp2f(-(float)(2 * p) * (LOG2_10000 / (float)D_));
  const float fr  = (float)s * inv;
  const float c  = cosf(fr);
  const float sn = sinf(fr);

  const size_t qo = ((size_t)(b * H_ + h) * S_ + s) * D_ + 2 * p;

  // q: scaled by D^-0.5 * log2(e) so scores live in exp2 space
  {
    float xr = (float)qkv[base], xi = (float)qkv[base + 1];
    qr[qo]     = (bf16_t)((xr * c - xi * sn) * QSCALE_);
    qr[qo + 1] = (bf16_t)((xr * sn + xi * c) * QSCALE_);
  }
  // k
  {
    float xr = (float)qkv[base + E_], xi = (float)qkv[base + E_ + 1];
    kr[qo]     = (bf16_t)(xr * c - xi * sn);
    kr[qo + 1] = (bf16_t)(xr * sn + xi * c);
  }
  // v, transposed to (d, t): makes both attention matmul operand loads contiguous
  {
    float v0 = (float)qkv[base + 2 * E_], v1 = (float)qkv[base + 2 * E_ + 1];
    const size_t vo = ((size_t)(b * H_ + h) * D_ + 2 * p) * S_ + s;
    vt[vo]      = (bf16_t)v0;
    vt[vo + S_] = (bf16_t)v1;
  }
}

// ---------------------------------------------------------------------------
// Kernel 3: flash attention, transposed-score formulation.
//   S^T = K x Q^T  -> lane owns one query column; softmax reductions are in-lane
//   ctx^T = V^T x P^T -> P^T is already B-fragment layout (4 dword swaps fix halves)
// One wave per 16-query tile; streams 32 keys/iteration; no LDS.
// ---------------------------------------------------------------------------
__global__ __launch_bounds__(128) void flash_attn_wmma(const bf16_t* __restrict__ qr,
                                                       const bf16_t* __restrict__ kr,
                                                       const bf16_t* __restrict__ vt,
                                                       bf16_t* __restrict__ ctxb) {
  const int wave = threadIdx.x >> 5;
  const int wid  = blockIdx.x * 4 + wave;       // 4096 waves total
  const int st   = wid & (S_ / 16 - 1);         // query tile, 128 per (b,h)
  const int bh   = wid >> 7;                    // b*H + h, 0..31
  const int lane = threadIdx.x & 31;
  const int half = lane >> 4;
  const int r    = lane & 15;
  const int s0   = st << 4;

  const bf16_t* qbase = qr + (size_t)bh * S_ * D_;
  const bf16_t* kbase = kr + (size_t)bh * S_ * D_;
  const bf16_t* vbase = vt + (size_t)bh * D_ * S_;

  // Q as loop-invariant B fragments (column = query r, K-dim = d)
  const bf16_t* qrow = qbase + (size_t)(s0 + r) * D_;
  const v16bf qb0 = load_b_frag(qrow, 0, half);
  const v16bf qb1 = load_b_frag(qrow, 32, half);

  // ctx^T accumulators: tile n holds d = 16n + e + 8*half, query = lane&15
  v8f ctx0 = {}, ctx1 = {}, ctx2 = {}, ctx3 = {};
  float rm = -3.0e38f;   // running max (exp2 space), one query per lane
  float rl = 0.0f;       // running denominator

  for (int t0 = 0; t0 < S_; t0 += 32) {
    // S^T tiles: st0 keys t0+e+8*half, st1 keys t0+16+e+8*half, query = r
    const bf16_t* k0row = kbase + (size_t)(t0 + r) * D_;
    const bf16_t* k1row = kbase + (size_t)(t0 + 16 + r) * D_;
    v8f st0 = {}, st1 = {};
    st0 = wmma_bf16(load_a_frag(k0row, 0, half),  qb0, st0);
    st0 = wmma_bf16(load_a_frag(k0row, 32, half), qb1, st0);
    st1 = wmma_bf16(load_a_frag(k1row, 0, half),  qb0, st1);
    st1 = wmma_bf16(load_a_frag(k1row, 32, half), qb1, st1);

    // block max over this lane's 16 keys, then one cross-half combine
    float lmax = fmaxf(st0[0], st1[0]);
#pragma unroll
    for (int e = 1; e < 8; ++e) lmax = fmaxf(lmax, fmaxf(st0[e], st1[e]));
    lmax = fmaxf(lmax, __shfl_xor(lmax, 16, 32));

    const float mnew  = fmaxf(rm, lmax);
    const float alpha = fast_exp2(rm - mnew);
    rm = mnew;

    float p0[8], p1[8];
    float ls = 0.0f;
#pragma unroll
    for (int e = 0; e < 8; ++e) {
      p0[e] = fast_exp2(st0[e] - mnew);
      p1[e] = fast_exp2(st1[e] - mnew);
      ls += p0[e] + p1[e];
    }
    ls += __shfl_xor(ls, 16, 32);
    rl = rl * alpha + ls;

#pragma unroll
    for (int e = 0; e < 8; ++e) {
      ctx0[e] *= alpha; ctx1[e] *= alpha; ctx2[e] *= alpha; ctx3[e] *= alpha;
    }

    // Build P^T B-fragment. B layout wants: elements j -> key t0 + 16*half + j.
    // half0 lane owns keys 0..7 (st0), partner (lane^16) owns keys 8..15 of st0;
    // half1 lane owns keys 24..31 (st1), partner owns keys 16..23 of st1.
    Frag16 pb;
#pragma unroll
    for (int j = 0; j < 4; ++j) {
      const uint own0 = pack2_bf16(p0[2 * j], p0[2 * j + 1]);   // tile0 pair
      const uint own1 = pack2_bf16(p1[2 * j], p1[2 * j + 1]);   // tile1 pair
      const uint send = half ? own0 : own1;
      const uint recv = __shfl_xor(send, 16, 32);
      pb.u[j]     = half ? recv : own0;   // keys 16*half + 0..7
      pb.u[4 + j] = half ? own1 : recv;   // keys 16*half + 8..15
    }

    // ctx^T tile n += V^T(16x32) x P^T(32x16); V^T rows are d, contiguous in t
    ctx0 = wmma_bf16(load_a_frag(vbase + (size_t)(0  + r) * S_, t0, half), pb.v, ctx0);
    ctx1 = wmma_bf16(load_a_frag(vbase + (size_t)(16 + r) * S_, t0, half), pb.v, ctx1);
    ctx2 = wmma_bf16(load_a_frag(vbase + (size_t)(32 + r) * S_, t0, half), pb.v, ctx2);
    ctx3 = wmma_bf16(load_a_frag(vbase + (size_t)(48 + r) * S_, t0, half), pb.v, ctx3);
  }

  // Normalize and store: lane owns query s0+r; tile n elements are d = 16n+8*half+0..7,
  // contiguous -> one b128 store per tile.
  const int b = bh >> 4;
  const int h = bh & 15;
  const float il = 1.0f / rl;
  bf16_t* o = ctxb + (size_t)(b * S_ + (s0 + r)) * E_ + h * D_;

  Pack8 w0, w1, w2, w3;
#pragma unroll
  for (int j = 0; j < 4; ++j) {
    w0.u[j] = pack2_bf16(ctx0[2 * j] * il, ctx0[2 * j + 1] * il);
    w1.u[j] = pack2_bf16(ctx1[2 * j] * il, ctx1[2 * j + 1] * il);
    w2.u[j] = pack2_bf16(ctx2[2 * j] * il, ctx2[2 * j + 1] * il);
    w3.u[j] = pack2_bf16(ctx3[2 * j] * il, ctx3[2 * j + 1] * il);
  }
  *(uint4*)(o + 0  + 8 * half) = w0.q;
  *(uint4*)(o + 16 + 8 * half) = w1.q;
  *(uint4*)(o + 32 + 8 * half) = w2.q;
  *(uint4*)(o + 48 + 8 * half) = w3.q;
}

// ---------------------------------------------------------------------------
// Host-side orchestration (all on `stream`; workspace carved from d_ws)
// ---------------------------------------------------------------------------
extern "C" void kernel_launch(void* const* d_in, const int* in_sizes, int n_in,
                              void* d_out, int out_size, void* d_ws, size_t ws_size,
                              hipStream_t stream) {
  (void)in_sizes; (void)n_in; (void)out_size; (void)ws_size;
  const float* query = (const float*)d_in[0];
  // d_in[1] (key) and d_in[2] (value) are unused by the reference: qkv comes from `query`.
  const float* w_qkv = (const float*)d_in[3];
  const float* w_out = (const float*)d_in[4];
  float* out = (float*)d_out;

  char* ws = (char*)d_ws;
  // Workspace layout (64 MB total)
  bf16_t* Xb    = (bf16_t*)(ws + 0);          //  8 MB  query bf16 (B*S, E)
  bf16_t* ctxb  = Xb;                         //  reuse: Xb consumed before ctx written
  bf16_t* Wqkvb = (bf16_t*)(ws + 8388608);    //  6 MB  (3E, E)
  bf16_t* Woutb = (bf16_t*)(ws + 14680064);   //  2 MB  (E, E)
  bf16_t* qkvb  = (bf16_t*)(ws + 16777216);   // 24 MB  (B*S, 3E)
  bf16_t* qrp   = (bf16_t*)(ws + 41943040);   //  8 MB  (B,H,S,D)
  bf16_t* krp   = (bf16_t*)(ws + 50331648);   //  8 MB  (B,H,S,D)
  bf16_t* vtp   = (bf16_t*)(ws + 58720256);   //  8 MB  (B,H,D,S)

  const int nQ    = B_ * S_ * E_;     // 4194304
  const int nWqkv = 3 * E_ * E_;      // 3145728
  const int nWout = E_ * E_;          // 1048576

  f32_to_bf16_k<<<(nQ + 255) / 256, 256, 0, stream>>>(query, Xb, nQ);
  f32_to_bf16_k<<<(nWqkv + 255) / 256, 256, 0, stream>>>(w_qkv, Wqkvb, nWqkv);
  f32_to_bf16_k<<<(nWout + 255) / 256, 256, 0, stream>>>(w_out, Woutb, nWout);

  // QKV projection: M=4096, N=3072, K=1024 -> 12288 waves -> 3072 blocks
  gemm_bt_wmma<true><<<3072, 128, 0, stream>>>(Xb, Wqkvb, qkvb, (float*)nullptr,
                                               B_ * S_, 3 * E_, E_);

  // RoPE + scatter: B*H*S*(D/2) = 2^21 threads
  rope_scatter_k<<<(B_ * H_ * S_ * (D_ / 2)) / 256, 256, 0, stream>>>(qkvb, qrp, krp, vtp);

  // Flash attention: B*H*(S/16) = 4096 waves -> 1024 blocks of 4 waves
  flash_attn_wmma<<<1024, 128, 0, stream>>>(qrp, krp, vtp, ctxb);

  // Output projection: M=4096, N=1024, K=1024 -> 4096 waves -> 1024 blocks
  gemm_bt_wmma<false><<<1024, 128, 0, stream>>>(ctxb, Woutb, (bf16_t*)nullptr, out,
                                                B_ * S_, E_, E_);
}
